// SparseLinearAttention_8461085573738
// MI455X (gfx1250) — matrix operand order repaired and checked by reading, not compiled, not run on hardware
//
#include <hip/hip_runtime.h>

// ---------------------------------------------------------------------------
// SparseLinearAttention for MI455X (gfx1250, wave32, WMMA f32_16x16x32_f16)
// B=2 H=16 L=4096 D=64, BLK=64, nQ=nK=64, T=6 selected key blocks
//
// v3b: k/v pre-converted to f16 in global memory once (K0); V stored
//      block-transposed [e][key].  The hot kernels fill their padded LDS tiles
//      with the Tensor Data Mover (tensor_load_to_lds, TENSORcnt), using the
//      TDM pad feature (4 DWORDs every 32 DWORDs) to produce the 72-half row
//      stride in hardware.  All WMMA fragments remain 2 x ds_load_b128.
// ---------------------------------------------------------------------------

typedef __attribute__((ext_vector_type(16))) _Float16 v16h;
typedef __attribute__((ext_vector_type(8)))  _Float16 v8h;
typedef __attribute__((ext_vector_type(2)))  __fp16   v2x;   // cvt_pkrtz result type
typedef __attribute__((ext_vector_type(8)))  float    v8f;
typedef __attribute__((ext_vector_type(4)))  unsigned int v4u;
typedef __attribute__((ext_vector_type(8)))  int      v8i;
typedef __attribute__((ext_vector_type(4)))  int      v4i;

#define BH_    32
#define L_     4096
#define D_     64
#define NB_    64
#define TK_    6
#define SCALE_ 0.125f   // 1/sqrt(64)
#define LDT_   72       // padded f16 tile stride (halves) == 32+4 DWORDs
#define LDF_   65       // padded f32 tile stride (floats)

#define WMMA(a, b, c) \
  __builtin_amdgcn_wmma_f32_16x16x32_f16(false, (a), false, (b), (short)0, (c), false, false)

static __device__ __forceinline__ v16h cat8(v8h lo, v8h hi) {
  return __builtin_shufflevector(lo, hi, 0, 1, 2, 3, 4, 5, 6, 7,
                                         8, 9, 10, 11, 12, 13, 14, 15);
}

// ---- WMMA fragment loaders (layouts per CDNA5 ISA 7.12.2, wave32) ----------
static __device__ __forceinline__ v16h ldA_rm(const _Float16* lds, int row0, int col0, int ld) {
  const int ln = threadIdx.x & 31;
  const _Float16* p = lds + (row0 + (ln & 15)) * ld + col0 + ((ln < 16) ? 0 : 8);
  const v8h lo = *(const v8h*)p;         // K = kb + 0..7
  const v8h hi = *(const v8h*)(p + 16);  // K = kb + 16..23
  return cat8(lo, hi);
}
static __device__ __forceinline__ v16h ldB_kc(const _Float16* lds, int k0, int n0, int ld) {
  const int ln = threadIdx.x & 31;
  const _Float16* p = lds + (n0 + (ln & 15)) * ld + k0 + ((ln < 16) ? 0 : 8);
  const v8h lo = *(const v8h*)p;         // K = kb + 0..7
  const v8h hi = *(const v8h*)(p + 16);  // K = kb + 16..23
  return cat8(lo, hi);
}
// C layout: VGPR j, lane ln -> M = (ln<16 ? j : j+8), N = ln&15

static __device__ __forceinline__ float hmax16(float x) {
#pragma unroll
  for (int m = 8; m >= 1; m >>= 1) x = fmaxf(x, __shfl_xor(x, m, 32));
  return x;
}
static __device__ __forceinline__ float hsum16(float x) {
#pragma unroll
  for (int m = 8; m >= 1; m >>= 1) x += __shfl_xor(x, m, 32);
  return x;
}
static __device__ __forceinline__ void zero8(v8f& a) {
#pragma unroll
  for (int i = 0; i < 8; ++i) a[i] = 0.0f;
}

// ---- Tensor Data Mover: contiguous 64x64 f16 tile -> padded LDS ------------
// D# per CDNA5 ISA ch.8: 2D tensor, data_size=2B, tile 64x64, row stride 64,
// pad_enable: pad_interval=4 (32 DWORDs = one 64-half row), pad_amount=3
// (4 DWORDs = 8 halves) -> LDS row stride 72 halves = LDT_.
static __device__ __forceinline__ void tdm_load_tile(const _Float16* g, void* lds) {
  const unsigned long long ga = (unsigned long long)(size_t)g;
  const unsigned la = (unsigned)(size_t)lds;  // flat LDS addr[31:0] = byte offset
  v4u g0;
  g0[0] = 1u;                                            // count=1, user mode
  g0[1] = la;                                            // lds_addr
  g0[2] = (unsigned)ga;                                  // global_addr[31:0]
  g0[3] = (unsigned)((ga >> 32) & 0x01FFFFFFu) | (2u << 30);  // addr[56:32], type=2
  v8i g1;
  g1[0] = (1 << 16) | (1 << 20) | (4 << 22) | (3 << 25); // data_size=2B, pad_en,
                                                         // pad_int=32dw, pad_amt=4dw
  g1[1] = (64 << 16);   // atomic_barrier=0 | tensor_dim0[15:0]=64
  g1[2] = (64 << 16);   // tensor_dim0[31:16]=0 | tensor_dim1[15:0]=64
  g1[3] = (64 << 16);   // tensor_dim1[31:16]=0 | tile_dim0=64
  g1[4] = 64;           // tile_dim1=64 | tile_dim2=0
  g1[5] = 64;           // tensor_dim0_stride[31:0]=64
  g1[6] = 0;            // stride[47:32]=0 | tensor_dim1_stride lo=0
  g1[7] = 0;
  v4i z4; z4[0] = 0; z4[1] = 0; z4[2] = 0; z4[3] = 0;
  v8i z8;
#pragma unroll
  for (int i = 0; i < 8; ++i) z8[i] = 0;
  __builtin_amdgcn_tensor_load_to_lds(g0, g1, z4, z4, z8, 0);
}

// 64x64 f32 global tile -> TRANSPOSED padded f16 LDS: lds[col][row]
static __device__ __forceinline__ void tile_f16_tr(const float* __restrict__ g,
                                                   _Float16* lds, int tid) {
  const float4* g4 = (const float4*)g;
#pragma unroll
  for (int i = 0; i < 8; ++i) {
    const int idx = tid + i * 128;
    const float4 x = g4[idx];
    const int row = idx >> 4;
    const int col = (idx & 15) * 4;
    lds[(col + 0) * LDT_ + row] = (_Float16)x.x;
    lds[(col + 1) * LDT_ + row] = (_Float16)x.y;
    lds[(col + 2) * LDT_ + row] = (_Float16)x.z;
    lds[(col + 3) * LDT_ + row] = (_Float16)x.w;
  }
}

// ---------------------------------------------------------------------------
// K0: k -> f16 row-major copy; v -> f16 block-transposed copy [blk][e][key]
// grid = BH*NB, block = 128
// ---------------------------------------------------------------------------
__global__ __launch_bounds__(128) void sla_cvt_kernel(
    const float* __restrict__ kg, const float* __restrict__ vg,
    _Float16* __restrict__ k16, _Float16* __restrict__ v16T) {
  __shared__ _Float16 vt[64 * LDT_];
  const int tid = threadIdx.x;
  const size_t gbase = (size_t)blockIdx.x * 4096;

  { // k: straight f32 -> f16 (packed converts, b64 stores)
    const float4* g4 = (const float4*)(kg + gbase);
    v2x* o = (v2x*)(k16 + gbase);
#pragma unroll
    for (int i = 0; i < 8; ++i) {
      const int idx = tid + i * 128;
      const float4 x = g4[idx];
      o[2 * idx]     = __builtin_amdgcn_cvt_pkrtz(x.x, x.y);
      o[2 * idx + 1] = __builtin_amdgcn_cvt_pkrtz(x.z, x.w);
    }
  }
  // v: transpose through LDS, then contiguous f16 writeback
  tile_f16_tr(vg + gbase, vt, tid);
  __syncthreads();
  {
    v8h* vo = (v8h*)(v16T + gbase);
#pragma unroll
    for (int i = 0; i < 4; ++i) {
      const int c = tid + i * 128;          // 512 chunks of 8 halves
      const int row = c >> 3;
      const int col = (c & 7) * 8;
      vo[c] = *(const v8h*)(vt + row * LDT_ + col);
    }
  }
}

// ---------------------------------------------------------------------------
// K1: block mean pooling of q and k.  grid = BH*NB, block = 64
// ---------------------------------------------------------------------------
__global__ __launch_bounds__(64) void sla_means_kernel(
    const float* __restrict__ q, const float* __restrict__ k,
    float* __restrict__ qm, float* __restrict__ km) {
  const int d = threadIdx.x;
  const size_t base = (size_t)blockIdx.x * 64 * D_;
  float sq = 0.f, sk = 0.f;
#pragma unroll 8
  for (int i = 0; i < 64; ++i) {
    sq += q[base + (size_t)i * D_ + d];
    sk += k[base + (size_t)i * D_ + d];
  }
  qm[(size_t)blockIdx.x * D_ + d] = sq * (1.0f / 64.0f);
  km[(size_t)blockIdx.x * D_ + d] = sk * (1.0f / 64.0f);
}

// ---------------------------------------------------------------------------
// K2: block scores + top-6 selection.  grid = BH*NB, block = 64
// ---------------------------------------------------------------------------
__global__ __launch_bounds__(64) void sla_topk_kernel(
    const float* __restrict__ qm, const float* __restrict__ km,
    int* __restrict__ lut) {
  __shared__ float sc[NB_];
  const int j  = threadIdx.x;
  const int bh = blockIdx.x >> 6;
  const float* qv = qm + (size_t)blockIdx.x * D_;
  const float* kb = km + (size_t)bh * NB_ * D_ + (size_t)j * D_;
  float s = 0.f;
#pragma unroll 8
  for (int d = 0; d < D_; ++d) s += qv[d] * kb[d];
  sc[j] = s * SCALE_;
  __syncthreads();
  if (j == 0) {
    int* lo = lut + (size_t)blockIdx.x * TK_;
    for (int t = 0; t < TK_; ++t) {
      float best = -1e30f; int bi = 0;
      for (int i = 0; i < NB_; ++i)
        if (sc[i] > best) { best = sc[i]; bi = i; }
      lo[t] = bi;
      sc[bi] = -1e30f;
    }
  }
}

// ---------------------------------------------------------------------------
// K3: per key block: ck = softmax(k,rows), kv = ck^T @ v (WMMA), z = colsum(ck)
// grid = BH*NB, block = 128.  vT tile filled by TDM from v16T.
// ---------------------------------------------------------------------------
__global__ __launch_bounds__(128) void sla_kvz_kernel(
    const float* __restrict__ kg, const _Float16* __restrict__ v16T,
    float* __restrict__ kv, float* __restrict__ z) {
  __shared__ float    kf[64 * LDF_];
  __shared__ _Float16 ckT[64 * LDT_];
  __shared__ _Float16 vT[64 * LDT_];
  const int tid = threadIdx.x;
  const int w   = tid >> 5;
  const int ln  = tid & 31;

  if (w == 0) tdm_load_tile(v16T + (size_t)blockIdx.x * 4096, vT);  // async

  { // coalesced f32 load of k tile -> padded f32 LDS (overlaps the TDM)
    const float4* g4 = (const float4*)(kg + (size_t)blockIdx.x * 4096);
#pragma unroll
    for (int i = 0; i < 8; ++i) {
      const int idx = tid + i * 128;
      const float4 x = g4[idx];
      float* p = kf + (idx >> 4) * LDF_ + (idx & 15) * 4;
      p[0] = x.x; p[1] = x.y; p[2] = x.z; p[3] = x.w;
    }
  }
  if (w == 0) __builtin_amdgcn_s_wait_tensorcnt(0);
  __syncthreads();

  if (tid < 64) {  // row softmax of k -> ckT[d][row]
    float mx = -1e30f;
#pragma unroll 8
    for (int d = 0; d < D_; ++d) mx = fmaxf(mx, kf[tid * LDF_ + d]);
    float s = 0.f;
#pragma unroll 8
    for (int d = 0; d < D_; ++d) s += __expf(kf[tid * LDF_ + d] - mx);
    const float inv = 1.0f / s;
#pragma unroll 8
    for (int d = 0; d < D_; ++d)
      ckT[d * LDT_ + tid] = (_Float16)(__expf(kf[tid * LDF_ + d] - mx) * inv);
  }
  __syncthreads();

  if (tid < 64) {  // z[d] = sum_m ck[m][d]  (contiguous in ckT)
    float s = 0.f;
#pragma unroll 8
    for (int m = 0; m < 64; ++m) s += (float)ckT[tid * LDT_ + m];
    z[(size_t)blockIdx.x * D_ + tid] = s;
  }

  // kv[d][e] = sum_m ck[m][d] * v[m][e]
  const v16h a0 = ldA_rm(ckT, 16 * w, 0, LDT_);
  const v16h a1 = ldA_rm(ckT, 16 * w, 32, LDT_);
#pragma unroll
  for (int n = 0; n < 4; ++n) {
    v8f acc; zero8(acc);
    acc = WMMA(a0, ldB_kc(vT, 0,  16 * n, LDT_), acc);
    acc = WMMA(a1, ldB_kc(vT, 32, 16 * n, LDT_), acc);
    const int e = 16 * n + (ln & 15);
#pragma unroll
    for (int j = 0; j < 8; ++j) {
      const int d = 16 * w + ((ln < 16) ? j : j + 8);
      kv[(size_t)blockIdx.x * 4096 + (size_t)d * 64 + e] = acc[j];
    }
  }
}

// ---------------------------------------------------------------------------
// K4: per-(bh) totals of kv and z over all key blocks.  grid = BH, block = 256
// ---------------------------------------------------------------------------
__global__ __launch_bounds__(256) void sla_totals_kernel(
    const float* __restrict__ kv, const float* __restrict__ z,
    float* __restrict__ kvt, float* __restrict__ zt) {
  const int bh  = blockIdx.x;
  const int tid = threadIdx.x;
  for (int i = tid; i < 4096; i += 256) {
    float s = 0.f;
#pragma unroll 8
    for (int kb = 0; kb < NB_; ++kb)
      s += kv[((size_t)bh * NB_ + kb) * 4096 + i];
    kvt[(size_t)bh * 4096 + i] = s;
  }
  if (tid < 64) {
    float s = 0.f;
#pragma unroll 8
    for (int kb = 0; kb < NB_; ++kb)
      s += z[((size_t)bh * NB_ + kb) * 64 + tid];
    zt[(size_t)bh * 64 + tid] = s;
  }
}

// ---------------------------------------------------------------------------
// K5: fused sparse flash attention + complement linear attention + projection
// grid = BH*NB, block = 128 (4 waves, 16 output rows each)
// K/V chunk tiles filled by TDM from the f16 global copies.
// ---------------------------------------------------------------------------
__global__ __launch_bounds__(128) void sla_attn_kernel(
    const float* __restrict__ q,  const _Float16* __restrict__ k16,
    const _Float16* __restrict__ v16T, const float* __restrict__ Wl,
    const float* __restrict__ bl, const int* __restrict__ lut,
    const float* __restrict__ kv, const float* __restrict__ z,
    const float* __restrict__ kvt, const float* __restrict__ zt,
    float* __restrict__ out) {
  __shared__ float    sf[64 * LDF_];                      // f32 q staging
  __shared__ _Float16 qh[64 * LDT_], cqh[64 * LDT_];      // scaled q, softmax(q)
  __shared__ _Float16 kh[64 * LDT_];                      // K chunk [key][d]; later W_l [e][d]
  __shared__ _Float16 vh[64 * LDT_];                      // V chunk [e][key]; later kv_ns [e][d]
  __shared__ _Float16 ph[64 * LDT_];                      // P / o_l staging (per-wave slices)
  __shared__ float    zns[64], den[64];

  const int tid = threadIdx.x;
  const int w   = tid >> 5;
  const int ln  = tid & 31;
  const int bh  = blockIdx.x >> 6;
  const int qb  = blockIdx.x & 63;
  const size_t qbase = (size_t)bh * L_ * D_ + (size_t)qb * 64 * D_;

  { // load q tile -> sf (f32) and qh (scaled f16, row-major)
    const float4* g4 = (const float4*)(q + qbase);
#pragma unroll
    for (int i = 0; i < 8; ++i) {
      const int idx = tid + i * 128;
      const float4 x = g4[idx];
      const int row = idx >> 4;
      const int col = (idx & 15) * 4;
      float* pf = sf + row * LDF_ + col;
      pf[0] = x.x; pf[1] = x.y; pf[2] = x.z; pf[3] = x.w;
      _Float16* p = qh + row * LDT_ + col;
      p[0] = (_Float16)(x.x * SCALE_); p[1] = (_Float16)(x.y * SCALE_);
      p[2] = (_Float16)(x.z * SCALE_); p[3] = (_Float16)(x.w * SCALE_);
    }
  }
  __syncthreads();
  if (tid < 64) {  // c_q = softmax(q, rows)
    float mx = -1e30f;
#pragma unroll 8
    for (int d = 0; d < D_; ++d) mx = fmaxf(mx, sf[tid * LDF_ + d]);
    float s = 0.f;
#pragma unroll 8
    for (int d = 0; d < D_; ++d) s += __expf(sf[tid * LDF_ + d] - mx);
    const float inv = 1.0f / s;
#pragma unroll 8
    for (int d = 0; d < D_; ++d)
      cqh[tid * LDT_ + d] = (_Float16)(__expf(sf[tid * LDF_ + d] - mx) * inv);
  }
  __syncthreads();

  // hoisted A fragments (q and c_q never change)
  const v16h aq0 = ldA_rm(qh,  16 * w, 0, LDT_);
  const v16h aq1 = ldA_rm(qh,  16 * w, 32, LDT_);
  const v16h ac0 = ldA_rm(cqh, 16 * w, 0, LDT_);
  const v16h ac1 = ldA_rm(cqh, 16 * w, 32, LDT_);

  float mrow[8], lrow[8];
  v8f oacc[4];
#pragma unroll
  for (int j = 0; j < 8; ++j) { mrow[j] = -1e30f; lrow[j] = 0.f; }
#pragma unroll
  for (int n = 0; n < 4; ++n) zero8(oacc[n]);

  const int* lutq = lut + (size_t)blockIdx.x * TK_;

  // ---- sparse flash attention over 6 selected 64-key chunks ----
  for (int t = 0; t < TK_; ++t) {
    const int kbi = lutq[t];
    __syncthreads();  // all waves done with previous chunk's LDS
    if (w == 0) {     // TDM fills both padded tiles; wave-level, EXEC-independent
      const size_t tb = ((size_t)bh * NB_ + kbi) * 4096;
      tdm_load_tile(k16  + tb, kh);   // [key][d]
      tdm_load_tile(v16T + tb, vh);   // [e][key]
      __builtin_amdgcn_s_wait_tensorcnt(0);
    }
    __syncthreads();

    // S = (q*scale) @ K^T : B(k=d,n=key) = kh[key][d] (K-contiguous)
    v8f sc[4];
#pragma unroll
    for (int n = 0; n < 4; ++n) {
      zero8(sc[n]);
      sc[n] = WMMA(aq0, ldB_kc(kh, 0,  16 * n, LDT_), sc[n]);
      sc[n] = WMMA(aq1, ldB_kc(kh, 32, 16 * n, LDT_), sc[n]);
    }
    // online softmax update (row = C-layout M; N across 16 lanes)
#pragma unroll
    for (int j = 0; j < 8; ++j) {
      float tm = fmaxf(fmaxf(sc[0][j], sc[1][j]), fmaxf(sc[2][j], sc[3][j]));
      tm = hmax16(tm);
      const float nm   = fmaxf(mrow[j], tm);
      const float corr = __expf(mrow[j] - nm);
      mrow[j] = nm;
      const int rloc = 16 * w + ((ln < 16) ? j : j + 8);
      float ps = 0.f;
#pragma unroll
      for (int n = 0; n < 4; ++n) {
        const float p = __expf(sc[n][j] - nm);
        ps += p;
        ph[rloc * LDT_ + n * 16 + (ln & 15)] = (_Float16)p;  // per-wave slice
      }
      ps = hsum16(ps);
      lrow[j] = lrow[j] * corr + ps;
#pragma unroll
      for (int n = 0; n < 4; ++n) oacc[n][j] *= corr;
    }
    // O += P @ V : B(k=key,n=e) = vh[e][key] (K-contiguous)
    const v16h ap0 = ldA_rm(ph, 16 * w, 0, LDT_);
    const v16h ap1 = ldA_rm(ph, 16 * w, 32, LDT_);
#pragma unroll
    for (int n = 0; n < 4; ++n) {
      oacc[n] = WMMA(ap0, ldB_kc(vh, 0,  16 * n, LDT_), oacc[n]);
      oacc[n] = WMMA(ap1, ldB_kc(vh, 32, 16 * n, LDT_), oacc[n]);
    }
  }
  __syncthreads();

  // ---- linear path: kv_ns^T -> vh ([e][d]), W_l -> kh ([e][d]) ----
  {
    const float* kvb = kvt + (size_t)bh * 4096;
#pragma unroll
    for (int i = 0; i < 32; ++i) {
      const int idx = tid + i * 128;   // flat (d,e): d = idx>>6, e = idx&63
      float s = kvb[idx];
#pragma unroll
      for (int t = 0; t < TK_; ++t)
        s -= kv[((size_t)bh * NB_ + lutq[t]) * 4096 + idx];
      vh[(idx & 63) * LDT_ + (idx >> 6)] = (_Float16)s;          // transposed
      kh[(idx >> 6) * LDT_ + (idx & 63)] = (_Float16)Wl[idx];    // row-major (e,d)
    }
    if (tid < 64) {
      float s = zt[(size_t)bh * 64 + tid];
#pragma unroll
      for (int t = 0; t < TK_; ++t)
        s -= z[((size_t)bh * NB_ + lutq[t]) * 64 + tid];
      zns[tid] = s;
    }
  }
  __syncthreads();

  // num = c_q @ kv_ns : B(k=d,n=e) = vh[e][d];  den = c_q . z_ns
  v8f nacc[4];
#pragma unroll
  for (int n = 0; n < 4; ++n) {
    zero8(nacc[n]);
    nacc[n] = WMMA(ac0, ldB_kc(vh, 0,  16 * n, LDT_), nacc[n]);
    nacc[n] = WMMA(ac1, ldB_kc(vh, 32, 16 * n, LDT_), nacc[n]);
  }
  if (tid < 64) {
    float s = 0.f;
#pragma unroll 8
    for (int d = 0; d < D_; ++d) s += (float)cqh[tid * LDT_ + d] * zns[d];
    den[tid] = s;
  }
  __syncthreads();

  // o_l -> per-wave ph slice (f16), then proj: B(k=d,n=e) = kh[e][d]
#pragma unroll
  for (int j = 0; j < 8; ++j) {
    const int rloc = 16 * w + ((ln < 16) ? j : j + 8);
    const float dd = den[rloc] + 1e-6f;
#pragma unroll
    for (int n = 0; n < 4; ++n)
      ph[rloc * LDT_ + n * 16 + (ln & 15)] = (_Float16)(nacc[n][j] / dd);
  }
  const v16h al0 = ldA_rm(ph, 16 * w, 0, LDT_);
  const v16h al1 = ldA_rm(ph, 16 * w, 32, LDT_);
  v8f pacc[4];
#pragma unroll
  for (int n = 0; n < 4; ++n) {
    zero8(pacc[n]);
    pacc[n] = WMMA(al0, ldB_kc(kh, 0,  16 * n, LDT_), pacc[n]);
    pacc[n] = WMMA(al1, ldB_kc(kh, 32, 16 * n, LDT_), pacc[n]);
  }

  // out = o_s/l + proj + b_l
#pragma unroll
  for (int n = 0; n < 4; ++n) {
    const int e  = 16 * n + (ln & 15);
    const float be = bl[e];
#pragma unroll
    for (int j = 0; j < 8; ++j) {
      const int rloc = 16 * w + ((ln < 16) ? j : j + 8);
      const float os = oacc[n][j] / lrow[j];
      out[qbase + (size_t)rloc * D_ + e] = os + pacc[n][j] + be;
    }
  }
}

// ---------------------------------------------------------------------------
extern "C" void kernel_launch(void* const* d_in, const int* in_sizes, int n_in,
                              void* d_out, int out_size, void* d_ws, size_t ws_size,
                              hipStream_t stream) {
  (void)in_sizes; (void)n_in; (void)out_size; (void)ws_size;
  const float* q  = (const float*)d_in[0];
  const float* k  = (const float*)d_in[1];
  const float* v  = (const float*)d_in[2];
  const float* Wl = (const float*)d_in[3];
  const float* bl = (const float*)d_in[4];
  float* out = (float*)d_out;

  // workspace layout (float-sized slots)
  float* ws   = (float*)d_ws;
  float* qm   = ws;                         // 131072
  float* km   = qm + 131072;                // 131072
  int*   lut  = (int*)(km + 131072);        // 12288
  float* kv   = (float*)(lut + 12288);      // 8388608
  float* z    = kv + 8388608;               // 131072
  float* kvt  = z + 131072;                 // 131072
  float* zt   = kvt + 131072;               // 2048
  _Float16* k16  = (_Float16*)(zt + 2048);  // 8388608 halves (16.7MB)
  _Float16* v16T = k16 + 8388608;           // 8388608 halves (16.7MB)

  const int nblk = BH_ * NB_;  // 2048

  sla_cvt_kernel<<<nblk, 128, 0, stream>>>(k, v, k16, v16T);
  sla_means_kernel<<<nblk, 64, 0, stream>>>(q, k, qm, km);
  sla_topk_kernel<<<nblk, 64, 0, stream>>>(qm, km, lut);
  sla_kvz_kernel<<<nblk, 128, 0, stream>>>(k, v16T, kv, z);
  sla_totals_kernel<<<BH_, 256, 0, stream>>>(kv, z, kvt, zt);
  sla_attn_kernel<<<nblk, 128, 0, stream>>>(q, k16, v16T, Wl, bl, lut, kv, z, kvt, zt, out);
}
